// KNN_84069689852354
// MI455X (gfx1250) — matrix operand order repaired and checked by reading, not compiled, not run on hardware
//
#include <hip/hip_runtime.h>
#include <hip/hip_bf16.h>

// ---------------------------------------------------------------------------
// KNN via WMMA GEMM on gfx1250 (wave32).
//   B=2048 queries, D=384 features, N=65536 train points, out = Y_train[argmin].
//   cross = x @ X_train^T via v_wmma_f32_16x16x32_f16, fp32 accumulation.
//   Each wave owns TWO 16-row A tiles so every B fragment read from LDS feeds
//   two WMMAs (halves LDS bandwidth per FLOP vs one tile per wave).
// ---------------------------------------------------------------------------

typedef __attribute__((ext_vector_type(16))) _Float16 v16h;
typedef __attribute__((ext_vector_type(8)))  float    v8f;
typedef __attribute__((ext_vector_type(4)))  int      v4i;

#define BQ      2048
#define NT      65536
#define DD      384
#define KCH     12          // 384 / 32 K-chunks per tile
#define NCHUNK  64          // N split into 64 chunks of 1024
#define TILES_PER_CHUNK 64  // 1024 / 16 n-tiles per chunk
#define FRAG_HALVES 512     // one 16x32 (or 32x16) f16 fragment = 32 lanes * 16 halves
#define TILE_HALVES (KCH * FRAG_HALVES)   // 6144 halves = 12 KB per B tile

#if __has_builtin(__builtin_amdgcn_global_load_async_to_lds_b128) && \
    __has_builtin(__builtin_amdgcn_s_wait_asynccnt)
#define USE_ASYNC_LDS 1
#define AS_GLOBAL __attribute__((address_space(1)))
#define AS_LDS    __attribute__((address_space(3)))
#endif

// ---------------------------------------------------------------------------
// Row sum-of-squares: one wave per row, 384 cols (12 per lane), shuffle reduce.
// ---------------------------------------------------------------------------
__global__ __launch_bounds__(256) void row_norms(const float* __restrict__ src,
                                                 float* __restrict__ dst, int rows) {
    int row  = blockIdx.x * 8 + (threadIdx.x >> 5);
    int lane = threadIdx.x & 31;
    if (row >= rows) return;
    const float* p = src + (size_t)row * DD;
    float s = 0.f;
#pragma unroll
    for (int j = 0; j < 12; ++j) { float v = p[lane + 32 * j]; s += v * v; }
#pragma unroll
    for (int o = 16; o >= 1; o >>= 1) s += __shfl_xor(s, o, 32);
    if (lane == 0) dst[row] = s;
}

// ---------------------------------------------------------------------------
// Pack x into the WMMA A-fragment lane layout (16-bit A 16x32, ISA 7.12.2):
//   lane L: M = L%16 ; half h: K = (L/16)*8 + (h<8 ? h : 8+h)
// ---------------------------------------------------------------------------
__global__ __launch_bounds__(256) void pack_a(const float* __restrict__ x,
                                              _Float16* __restrict__ pa) {
    int tid = blockIdx.x * 256 + threadIdx.x;       // < 2048*384
    int h    = tid & 15;
    int r    = tid >> 4;
    int lane = r & 31;  r >>= 5;
    int kc   = r % KCH;
    int mt   = r / KCH;
    int k = kc * 32 + ((lane >> 4) * 8) + (h < 8 ? h : 8 + h);
    int m = mt * 16 + (lane & 15);
    pa[tid] = (_Float16)x[(size_t)m * DD + k];
}

// ---------------------------------------------------------------------------
// Pack X_train into the WMMA B-fragment lane layout (16-bit B 32x16):
//   lane L: N = L%16 ; half h: K = (L/16)*8 + (h<8 ? h : 8+h)
// ---------------------------------------------------------------------------
__global__ __launch_bounds__(256) void pack_b(const float* __restrict__ xt,
                                              _Float16* __restrict__ pb) {
    int tid = blockIdx.x * 256 + threadIdx.x;       // < 65536*384
    int h    = tid & 15;
    int r    = tid >> 4;
    int lane = r & 31;  r >>= 5;
    int kc   = r % KCH;
    int nt   = r / KCH;
    int k = kc * 32 + ((lane >> 4) * 8) + (h < 8 ? h : 8 + h);
    int n = nt * 16 + (lane & 15);
    pb[tid] = (_Float16)xt[(size_t)n * DD + k];
}

// ---------------------------------------------------------------------------
// Stage one 12 KB B tile global -> LDS (async on CDNA5, fallback plain copy).
// 256 threads * 16 B * 3 iters = 12288 B.
// ---------------------------------------------------------------------------
__device__ __forceinline__ void stage_tile(const _Float16* __restrict__ src,
                                           _Float16* dst, int tid) {
#ifdef USE_ASYNC_LDS
    AS_GLOBAL v4i* g = (AS_GLOBAL v4i*)src;
    AS_LDS    v4i* l = (AS_LDS    v4i*)dst;
#pragma unroll
    for (int i = 0; i < 3; ++i)
        __builtin_amdgcn_global_load_async_to_lds_b128(g + tid + i * 256,
                                                       l + tid + i * 256, 0, 0);
#else
    const float4* gs = (const float4*)src;
    float4*       ld = (float4*)dst;
#pragma unroll
    for (int i = 0; i < 3; ++i) ld[tid + i * 256] = gs[tid + i * 256];
#endif
}

__device__ __forceinline__ void stage_wait() {
#ifdef USE_ASYNC_LDS
    __builtin_amdgcn_s_wait_asynccnt(0);
#endif
}

// ---------------------------------------------------------------------------
// GEMM + running argmin.  grid = (8 M-groups, 64 N-chunks), 256 thr = 8 waves.
// Wave w owns query rows [ (blockIdx.x*16 + 2w)*16 , +32 ): two A tiles
// resident in VGPRs (24 fragments, 192 VGPRs).  Each B fragment from LDS
// feeds two WMMAs.
// ---------------------------------------------------------------------------
__global__ __launch_bounds__(256, 1) void gemm_argmin(
    const _Float16* __restrict__ pA, const _Float16* __restrict__ pB,
    const float* __restrict__ xx, const float* __restrict__ tt,
    float* __restrict__ wmin, int* __restrict__ widx) {
    __shared__ __attribute__((aligned(16))) _Float16 ldsB[2][TILE_HALVES];

    const int tid    = threadIdx.x;
    const int lane   = tid & 31;
    const int wave   = tid >> 5;
    const int mtile0 = blockIdx.x * 16 + wave * 2;   // two 16-row tiles per wave
    const int chunk  = blockIdx.y;
    const int t0     = chunk * TILES_PER_CHUNK;

    // Resident A tiles: 2 x 12 fragments (192 VGPRs of halves).
    v16h a[2][KCH];
#pragma unroll
    for (int ms = 0; ms < 2; ++ms)
#pragma unroll
        for (int k = 0; k < KCH; ++k)
            a[ms][k] = *(const v16h*)(pA +
                (((size_t)(mtile0 + ms) * KCH + k) * 32 + lane) * 16);

    // ||x||^2 for the 16 rows this lane's C registers cover.
    float xxv[2][8];
#pragma unroll
    for (int ms = 0; ms < 2; ++ms)
#pragma unroll
        for (int g = 0; g < 8; ++g)
            xxv[ms][g] = xx[(mtile0 + ms) * 16 + (lane >> 4) * 8 + g];

    float bestv[2][8];
    int   besti[2][8];
#pragma unroll
    for (int ms = 0; ms < 2; ++ms)
#pragma unroll
        for (int g = 0; g < 8; ++g) { bestv[ms][g] = 3.4e38f; besti[ms][g] = 0; }

    stage_tile(pB + (size_t)t0 * TILE_HALVES, &ldsB[0][0], tid);

    for (int t = 0; t < TILES_PER_CHUNK; ++t) {
        stage_wait();          // tile t landed in LDS (per-wave async ops)
        __syncthreads();       // ...for all waves; prev buffer free for reuse
        if (t + 1 < TILES_PER_CHUNK)
            stage_tile(pB + (size_t)(t0 + t + 1) * TILE_HALVES,
                       &ldsB[(t + 1) & 1][0], tid);

        const _Float16* bbuf = &ldsB[t & 1][0];
        v8f c0 = {}, c1 = {};
#pragma unroll
        for (int k = 0; k < KCH; ++k) {
            v16h b = *(const v16h*)(bbuf + k * FRAG_HALVES + lane * 16);
            c0 = __builtin_amdgcn_wmma_f32_16x16x32_f16(
                     false, a[0][k], false, b, (short)0, c0, false, false);
            c1 = __builtin_amdgcn_wmma_f32_16x16x32_f16(
                     false, a[1][k], false, b, (short)0, c1, false, false);
        }

        const int   n   = (t0 + t) * 16 + (lane & 15);
        const float ttv = tt[n];
#pragma unroll
        for (int g = 0; g < 8; ++g) {
            float d0 = xxv[0][g] + ttv - 2.0f * c0[g];
            float d1 = xxv[1][g] + ttv - 2.0f * c1[g];
            if (d0 < bestv[0][g]) { bestv[0][g] = d0; besti[0][g] = n; }
            if (d1 < bestv[1][g]) { bestv[1][g] = d1; besti[1][g] = n; }
        }
    }

    // Cross-lane argmin over the 16 lanes holding each row's N values.
#pragma unroll
    for (int ms = 0; ms < 2; ++ms)
#pragma unroll
        for (int g = 0; g < 8; ++g) {
            float v = bestv[ms][g];
            int   i = besti[ms][g];
#pragma unroll
            for (int s = 8; s >= 1; s >>= 1) {
                float ov = __shfl_xor(v, s, 32);
                int   oi = __shfl_xor(i, s, 32);
                if (ov < v || (ov == v && oi < i)) { v = ov; i = oi; }
            }
            if ((lane & 15) == 0) {
                int m = (mtile0 + ms) * 16 + (lane >> 4) * 8 + g;
                wmin[(size_t)m * NCHUNK + chunk] = v;
                widx[(size_t)m * NCHUNK + chunk] = i;
            }
        }
}

// ---------------------------------------------------------------------------
// Reduce 64 chunk candidates per row (2 per lane), gather Y_train[best][0..23].
// ---------------------------------------------------------------------------
__global__ __launch_bounds__(256) void finalize(const float* __restrict__ wmin,
                                                const int* __restrict__ widx,
                                                const float* __restrict__ Y,
                                                float* __restrict__ out) {
    int row  = blockIdx.x * 8 + (threadIdx.x >> 5);
    int lane = threadIdx.x & 31;
    const float* wm = wmin + (size_t)row * NCHUNK;
    const int*   wi = widx + (size_t)row * NCHUNK;
    float v = wm[lane];
    int   i = wi[lane];
    {
        float v2 = wm[lane + 32];
        int   i2 = wi[lane + 32];
        if (v2 < v || (v2 == v && i2 < i)) { v = v2; i = i2; }
    }
#pragma unroll
    for (int s = 16; s >= 1; s >>= 1) {
        float ov = __shfl_xor(v, s, 32);
        int   oi = __shfl_xor(i, s, 32);
        if (ov < v || (ov == v && oi < i)) { v = ov; i = oi; }
    }
    i = __shfl(i, 0, 32);
    if (lane < 24) out[(size_t)row * 24 + lane] = Y[(size_t)i * 24 + lane];
}

// ---------------------------------------------------------------------------
extern "C" void kernel_launch(void* const* d_in, const int* in_sizes, int n_in,
                              void* d_out, int out_size, void* d_ws, size_t ws_size,
                              hipStream_t stream) {
    (void)in_sizes; (void)n_in; (void)out_size; (void)ws_size;
    const float* x  = (const float*)d_in[0];   // [2048, 24, 16]
    const float* Xt = (const float*)d_in[1];   // [65536, 384]
    const float* Yt = (const float*)d_in[2];   // [65536, 24, 1]
    float* out = (float*)d_out;                // [2048, 24, 1]

    char*  ws  = (char*)d_ws;
    size_t off = 0;
    auto carve = [&](size_t bytes) {
        void* p = ws + off;
        off = (off + bytes + 255) & ~(size_t)255;
        return p;
    };
    _Float16* pA   = (_Float16*)carve((size_t)BQ * DD * 2);        // 1.5 MB
    _Float16* pB   = (_Float16*)carve((size_t)NT * DD * 2);        // 50.3 MB
    float*    xx   = (float*)carve((size_t)BQ * 4);
    float*    tt   = (float*)carve((size_t)NT * 4);
    float*    wmin = (float*)carve((size_t)BQ * NCHUNK * 4);
    int*      widx = (int*)carve((size_t)BQ * NCHUNK * 4);

    row_norms<<<BQ / 8, 256, 0, stream>>>(x, xx, BQ);
    row_norms<<<NT / 8, 256, 0, stream>>>(Xt, tt, NT);
    pack_a<<<(BQ * DD) / 256, 256, 0, stream>>>(x, pA);
    pack_b<<<(NT * DD) / 256, 256, 0, stream>>>(Xt, pB);
    gemm_argmin<<<dim3(BQ / 256, NCHUNK), 256, 0, stream>>>(pA, pB, xx, tt, wmin, widx);
    finalize<<<BQ / 8, 256, 0, stream>>>(wmin, widx, Yt, out);
}
